// MyGATv2_28441273434344
// MI455X (gfx1250) — matrix operand order repaired and checked by reading, not compiled, not run on hardware
//
#include <hip/hip_runtime.h>

#define LEAK 0.2f
#define EPSV 1e-9f

typedef __attribute__((ext_vector_type(16))) __bf16 v16bf;
typedef __attribute__((ext_vector_type(8)))  __bf16 v8bf;
typedef __attribute__((ext_vector_type(8)))  float  v8f;

// ---------- helpers ----------
__device__ __forceinline__ __bf16 f2bf(float f) {
  unsigned u = __float_as_uint(f);
  u = u + 0x7fffu + ((u >> 16) & 1u);            // round-to-nearest-even
  unsigned short s = (unsigned short)(u >> 16);
  return __builtin_bit_cast(__bf16, s);
}
// order-preserving float <-> u32 key (for atomicMax-based segment max)
__device__ __forceinline__ unsigned f2ord(float f) {
  unsigned u = __float_as_uint(f);
  return (u & 0x80000000u) ? ~u : (u | 0x80000000u);
}
__device__ __forceinline__ float ord2f(unsigned k) {
  unsigned u = (k & 0x80000000u) ? (k & 0x7fffffffu) : ~k;
  return __uint_as_float(u);
}
__device__ __forceinline__ bool finitef(float f) {
  return (__float_as_uint(f) & 0x7f800000u) != 0x7f800000u;
}

// ---------- operand conversion (one-time, out of the hot loop) ----------
// f32 [rows x K] row-major -> bf16 [rows x Kp] row-major, zero-padded
__global__ void k_conv_row(const float* __restrict__ A, __bf16* __restrict__ O,
                           int rows, int K, int Kp) {
  int t = blockIdx.x * blockDim.x + threadIdx.x;
  if (t >= rows * Kp) return;
  int r = t / Kp, k = t - r * Kp;
  float v = (k < K) ? A[(size_t)r * K + k] : 0.0f;
  O[t] = f2bf(v);
}
// f32 W [K x cols] -> bf16 Wt [cols x Kp] (transpose + pad)
__global__ void k_conv_tr(const float* __restrict__ W, __bf16* __restrict__ O,
                          int K, int cols, int Kp) {
  int t = blockIdx.x * blockDim.x + threadIdx.x;
  if (t >= cols * Kp) return;
  int c = t / Kp, k = t - c * Kp;
  float v = (k < K) ? W[(size_t)k * cols + c] : 0.0f;
  O[t] = f2bf(v);
}

// ---------- guard-free WMMA GEMM ----------
// Y[rows x cols] = act(A @ W + bias); A: bf16 [rows x Kp], Bt: bf16 [cols x Kp] (=W^T)
// Preconditions: rows%16==0, cols%16==0, Kp%32==0. Block = (cols/16) waves,
// wave w owns the 16x16 tile (blockIdx.x*16, w*16). EXEC stays all-ones.
__global__ void k_gemm_wmma(const __bf16* __restrict__ A, const __bf16* __restrict__ Bt,
                            const float* __restrict__ bias, float* __restrict__ Y,
                            int Kp, int cols, int act) {
  const int lane = threadIdx.x & 31;
  const int wave = threadIdx.x >> 5;
  const int half = lane >> 4;                     // 0 | 1
  const int l16  = lane & 15;
  const int m = blockIdx.x * 16 + l16;            // A row for this lane
  const int n = wave * 16 + l16;                  // B col for this lane

  // A layout: lane holds k = half*16 + j, j=0..15
  //   -> two contiguous 8-element runs at +half*16 and +half*16+8
  // B layout: lane holds k = half*16 + j, j=0..15 -> one contiguous 16-element run
  const __bf16* pa = A  + (size_t)m * Kp + half * 16;
  const __bf16* pb = Bt + (size_t)n * Kp + half * 16;

  union { v16bf v; v8bf h[2]; } a;
  v8f acc = {};
  for (int k0 = 0; k0 < Kp; k0 += 32) {
    a.h[0] = *(const v8bf*)(pa + k0);             // 16B load
    a.h[1] = *(const v8bf*)(pa + k0 + 8);         // 16B load
    v16bf b = *(const v16bf*)(pb + k0);           // 32B load
    acc = __builtin_amdgcn_wmma_f32_16x16x32_bf16(false, a.v, false, b, (short)0, acc,
                                                  false, false);
  }

  const float bv = bias[n];
  float* yp = Y + (size_t)(blockIdx.x * 16 + half * 8) * cols + n;
#pragma unroll
  for (int r = 0; r < 8; ++r) {                   // C/D: VGPR r -> row r (+8 hi half)
    float v = acc[r] + bv;
    if (act) v = fmaxf(v, 0.0f);
    yp[(size_t)r * cols] = v;
  }
}

// ---------- misc elementwise / fill ----------
__global__ void k_fill(unsigned* __restrict__ p, unsigned v, size_t n) {
  size_t t = (size_t)blockIdx.x * blockDim.x + threadIdx.x;
  if (t < n) p[t] = v;
}
__global__ void k_resid_relu(const float* __restrict__ agg, const float* __restrict__ res,
                             float* __restrict__ h, size_t n) {
  size_t t = (size_t)blockIdx.x * blockDim.x + threadIdx.x;
  if (t >= n) return;
  h[t] = fmaxf(agg[t] + res[t], 0.0f);
}

// ---------- edge pipeline (GATv2 attention); thread = (edge, head) ----------
__global__ void k_edge_logits(const float* __restrict__ el, const float* __restrict__ er,
                              const int* __restrict__ src, const int* __restrict__ dst,
                              const float* __restrict__ attn, float* __restrict__ z,
                              unsigned* __restrict__ nmax, int E) {
  int t = blockIdx.x * blockDim.x + threadIdx.x;
  if (t >= E * 4) return;
  int e = t >> 2, hh = t & 3;
  int s = src[e], d = dst[e];
  const float4* pl = (const float4*)(el + (size_t)s * 128 + hh * 32);
  const float4* pr = (const float4*)(er + (size_t)d * 128 + hh * 32);
  const float4* pa = (const float4*)(attn + hh * 32);
  float acc = 0.0f;
#pragma unroll
  for (int i = 0; i < 8; ++i) {
    float4 a = pl[i], b = pr[i], w = pa[i];
    float v0 = a.x + b.x; v0 = (v0 > 0.f) ? v0 : LEAK * v0;
    float v1 = a.y + b.y; v1 = (v1 > 0.f) ? v1 : LEAK * v1;
    float v2 = a.z + b.z; v2 = (v2 > 0.f) ? v2 : LEAK * v2;
    float v3 = a.w + b.w; v3 = (v3 > 0.f) ? v3 : LEAK * v3;
    acc += v0 * w.x + v1 * w.y + v2 * w.z + v3 * w.w;
  }
  z[t] = acc;
  atomicMax(&nmax[(size_t)d * 4 + hh], f2ord(acc));
}

__global__ void k_edge_exp(const int* __restrict__ dst, float* __restrict__ z,
                           const unsigned* __restrict__ nmax, float* __restrict__ nsum, int E) {
  int t = blockIdx.x * blockDim.x + threadIdx.x;
  if (t >= E * 4) return;
  int e = t >> 2, hh = t & 3;
  int d = dst[e];
  float m = ord2f(nmax[(size_t)d * 4 + hh]);
  if (!finitef(m)) m = 0.0f;
  float ex = __expf(z[t] - m);
  z[t] = ex;
  atomicAdd(&nsum[(size_t)d * 4 + hh], ex);
}

__global__ void k_edge_msg(const int* __restrict__ src, const int* __restrict__ dst,
                           const float* __restrict__ el, const float* __restrict__ z,
                           const float* __restrict__ nsum, float* __restrict__ agg, int E) {
  int t = blockIdx.x * blockDim.x + threadIdx.x;
  if (t >= E * 4) return;
  int e = t >> 2, hh = t & 3;
  int s = src[e], d = dst[e];
  float alpha = z[t] / (nsum[(size_t)d * 4 + hh] + EPSV);
  const float4* pl = (const float4*)(el + (size_t)s * 128 + hh * 32);
  float* pd = agg + (size_t)d * 128 + hh * 32;
#pragma unroll
  for (int i = 0; i < 8; ++i) {
    float4 a = pl[i];
    atomicAdd(&pd[4 * i + 0], a.x * alpha);
    atomicAdd(&pd[4 * i + 1], a.y * alpha);
    atomicAdd(&pd[4 * i + 2], a.z * alpha);
    atomicAdd(&pd[4 * i + 3], a.w * alpha);
  }
}

// ---------- generic segment softmax (over nodes, per channel) ----------
__global__ void k_seg_max(const float* __restrict__ h, const int* __restrict__ seg,
                          unsigned* __restrict__ gmax, int N, int C) {
  int t = blockIdx.x * blockDim.x + threadIdx.x;
  if (t >= N * C) return;
  int n = t / C, c = t - n * C;
  atomicMax(&gmax[(size_t)seg[n] * C + c], f2ord(h[t]));
}
__global__ void k_seg_exp(const float* __restrict__ h, const int* __restrict__ seg,
                          const unsigned* __restrict__ gmax, float* __restrict__ gsum,
                          float* __restrict__ out, int N, int C) {
  int t = blockIdx.x * blockDim.x + threadIdx.x;
  if (t >= N * C) return;
  int n = t / C, c = t - n * C;
  float m = ord2f(gmax[(size_t)seg[n] * C + c]);
  if (!finitef(m)) m = 0.0f;
  float ex = __expf(h[t] - m);
  out[t] = ex;
  atomicAdd(&gsum[(size_t)seg[n] * C + c], ex);
}
__global__ void k_seg_norm(const int* __restrict__ seg, const float* __restrict__ gsum,
                           float* __restrict__ out, int N, int C) {
  int t = blockIdx.x * blockDim.x + threadIdx.x;
  if (t >= N * C) return;
  int n = t / C, c = t - n * C;
  out[t] = out[t] / (gsum[(size_t)seg[n] * C + c] + EPSV);
}

// ---------- gate + pooling ----------
__global__ void k_gate_logit(const float* __restrict__ newh, const float* __restrict__ gw,
                             const float* __restrict__ gb, float* __restrict__ gl, int N) {
  int n = blockIdx.x * blockDim.x + threadIdx.x;
  if (n >= N) return;
  const float4* p = (const float4*)(newh + (size_t)n * 128);
  const float4* w = (const float4*)gw;
  float acc = gb[0];
#pragma unroll
  for (int i = 0; i < 32; ++i) {
    float4 a = p[i], b = w[i];
    acc += a.x * b.x + a.y * b.y + a.z * b.z + a.w * b.w;
  }
  gl[n] = acc;
}
__global__ void k_pool(const float* __restrict__ newh, const float* __restrict__ gate,
                       const int* __restrict__ seg, float* __restrict__ hg, int N) {
  int t = blockIdx.x * blockDim.x + threadIdx.x;
  if (t >= N * 128) return;
  int n = t >> 7, c = t & 127;
  atomicAdd(&hg[(size_t)seg[n] * 128 + c], newh[t] * gate[n]);
}

// ---------- final classifier: out[B x 2] = z2[B x 64] @ cls_w[64 x 2] + cls_b ----------
__global__ void k_cls(const float* __restrict__ z2, const float* __restrict__ cw,
                      const float* __restrict__ cb, float* __restrict__ out, int B) {
  int t = blockIdx.x * blockDim.x + threadIdx.x;
  if (t >= B * 2) return;
  int b = t >> 1, c = t & 1;
  float acc = cb[c];
#pragma unroll 8
  for (int k = 0; k < 64; ++k) acc += z2[b * 64 + k] * cw[k * 2 + c];
  out[t] = acc;
}

// ---------- host side ----------
static inline void gemm(const __bf16* A, const __bf16* Bt, const float* bias, float* Y,
                        int rows, int Kp, int cols, int act, hipStream_t s) {
  dim3 block(32 * (cols / 16));                  // <= 256
  dim3 grid(rows / 16);
  hipLaunchKernelGGL(k_gemm_wmma, grid, block, 0, s, A, Bt, bias, Y, Kp, cols, act);
}
static inline void conv_row(const float* A, __bf16* O, int rows, int K, int Kp, hipStream_t s) {
  size_t n = (size_t)rows * Kp;
  hipLaunchKernelGGL(k_conv_row, dim3((n + 255) / 256), dim3(256), 0, s, A, O, rows, K, Kp);
}
static inline void conv_tr(const float* W, __bf16* O, int K, int cols, int Kp, hipStream_t s) {
  size_t n = (size_t)cols * Kp;
  hipLaunchKernelGGL(k_conv_tr, dim3((n + 255) / 256), dim3(256), 0, s, W, O, K, cols, Kp);
}
static inline void fill0(void* p, size_t nwords, hipStream_t s) {
  hipLaunchKernelGGL(k_fill, dim3((nwords + 255) / 256), dim3(256), 0, s, (unsigned*)p, 0u, nwords);
}

extern "C" void kernel_launch(void* const* d_in, const int* in_sizes, int n_in,
                              void* d_out, int out_size, void* d_ws, size_t ws_size,
                              hipStream_t stream) {
  const float* x      = (const float*)d_in[0];
  const int*   src    = (const int*)  d_in[1];
  const int*   dst    = (const int*)  d_in[2];
  const int*   seg    = (const int*)  d_in[3];
  const float* W0s    = (const float*)d_in[4];
  const float* b0s    = (const float*)d_in[5];
  const float* W0d    = (const float*)d_in[6];
  const float* b0d    = (const float*)d_in[7];
  const float* attn0  = (const float*)d_in[8];
  const float* Wres   = (const float*)d_in[9];
  const float* bres   = (const float*)d_in[10];
  const float* Ws     = (const float*)d_in[11];  // [2,128,128]
  const float* bs     = (const float*)d_in[12];  // [2,128]
  const float* Wd     = (const float*)d_in[13];
  const float* bd     = (const float*)d_in[14];
  const float* attn   = (const float*)d_in[15];  // [2,4,32]
  const float* gate_w = (const float*)d_in[16];
  const float* gate_b = (const float*)d_in[17];
  const float* lin_w  = (const float*)d_in[18];  // [128,64]
  const float* lin_b  = (const float*)d_in[19];
  const float* cls_w  = (const float*)d_in[20];  // [64,2]
  const float* cls_b  = (const float*)d_in[21];

  const int N = in_sizes[0] / 15;                // 100000 (multiple of 16)
  const int E = in_sizes[1];
  const int B = out_size / 2;                    // 64

  // workspace layout (f32 words)
  float* ws = (float*)d_ws;
  size_t o = 0;
  float*    h    = ws + o; o += (size_t)N * 128;
  float*    el   = ws + o; o += (size_t)N * 128;
  float*    er   = ws + o; o += (size_t)N * 128;
  float*    agg  = ws + o; o += (size_t)N * 128;
  float*    nres = ws + o; o += (size_t)N * 128;  // res (layer0), then newh
  __bf16*   Ab   = (__bf16*)(ws + o); o += (size_t)N * 64;  // bf16 activations [N x <=128]
  __bf16*   Wt1  = (__bf16*)(ws + o); o += 8192;            // bf16 weights [128 x 128]
  __bf16*   Wt2  = (__bf16*)(ws + o); o += 8192;
  float*    z    = ws + o; o += (size_t)E * 4;
  unsigned* nmax = (unsigned*)(ws + o); o += (size_t)N * 4;
  float*    nsum = ws + o; o += (size_t)N * 4;
  unsigned* gmax = (unsigned*)(ws + o); o += (size_t)B * 128;
  float*    gsum = ws + o; o += (size_t)B * 128;
  float*    glog = ws + o; o += (size_t)N;
  unsigned* gtmx = (unsigned*)(ws + o); o += (size_t)B;
  float*    gtsm = ws + o; o += (size_t)B;
  float*    hg   = ws + o; o += (size_t)B * 128;
  float*    z2   = ws + o; o += (size_t)64 * 64;

  const int TB = 256;
  const int gE  = (E * 4 + TB - 1) / TB;
  const int gNC = (N * 128 + TB - 1) / TB;
  const int gN1 = (N + TB - 1) / TB;

  fill0(hg, (size_t)B * 128, stream);

  // runs attention + pooling for one layer; Ab already holds bf16 input [N x Kp]
  auto run_layer = [&](int Kp, const float* Wsl, const float* bsl,
                       const float* Wdl, const float* bdl, const float* attnL,
                       const float* resbuf, int Kin) {
    conv_tr(Wsl, Wt1, Kin, 128, Kp, stream);
    gemm(Ab, Wt1, bsl, el, N, Kp, 128, 0, stream);           // fc_src (WMMA)
    conv_tr(Wdl, Wt2, Kin, 128, Kp, stream);
    gemm(Ab, Wt2, bdl, er, N, Kp, 128, 0, stream);           // fc_dst (WMMA)
    fill0(nmax, (size_t)N * 4, stream);
    fill0(nsum, (size_t)N * 4, stream);
    fill0(agg,  (size_t)N * 128, stream);
    hipLaunchKernelGGL(k_edge_logits, dim3(gE), dim3(TB), 0, stream,
                       el, er, src, dst, attnL, z, nmax, E);
    hipLaunchKernelGGL(k_edge_exp, dim3(gE), dim3(TB), 0, stream, dst, z, nmax, nsum, E);
    hipLaunchKernelGGL(k_edge_msg, dim3(gE), dim3(TB), 0, stream,
                       src, dst, el, z, nsum, agg, E);
    hipLaunchKernelGGL(k_resid_relu, dim3(gNC), dim3(TB), 0, stream,
                       agg, resbuf, h, (size_t)N * 128);
    // softmax_nodes over h -> newh (nres)
    fill0(gmax, (size_t)B * 128, stream);
    fill0(gsum, (size_t)B * 128, stream);
    hipLaunchKernelGGL(k_seg_max,  dim3(gNC), dim3(TB), 0, stream, h, seg, gmax, N, 128);
    hipLaunchKernelGGL(k_seg_exp,  dim3(gNC), dim3(TB), 0, stream, h, seg, gmax, gsum, nres, N, 128);
    hipLaunchKernelGGL(k_seg_norm, dim3(gNC), dim3(TB), 0, stream, seg, gsum, nres, N, 128);
    // gate softmax
    hipLaunchKernelGGL(k_gate_logit, dim3(gN1), dim3(TB), 0, stream, nres, gate_w, gate_b, glog, N);
    fill0(gtmx, (size_t)B, stream);
    fill0(gtsm, (size_t)B, stream);
    hipLaunchKernelGGL(k_seg_max,  dim3(gN1), dim3(TB), 0, stream, glog, seg, gtmx, N, 1);
    hipLaunchKernelGGL(k_seg_exp,  dim3(gN1), dim3(TB), 0, stream, glog, seg, gtmx, gtsm, glog, N, 1);
    hipLaunchKernelGGL(k_seg_norm, dim3(gN1), dim3(TB), 0, stream, seg, gtsm, glog, N, 1);
    hipLaunchKernelGGL(k_pool, dim3(gNC), dim3(TB), 0, stream, nres, glog, seg, hg, N);
  };

  // layer 0: 15 -> 128 (Kp=32), residual = x @ Wres + bres
  conv_row(x, Ab, N, 15, 32, stream);
  conv_tr(Wres, Wt1, 15, 128, 32, stream);
  gemm(Ab, Wt1, bres, nres, N, 32, 128, 0, stream);          // residual proj (WMMA)
  run_layer(32, W0s, b0s, W0d, b0d, attn0, nres, 15);
  // layers 1..2: 128 -> 128, identity residual
  conv_row(h, Ab, N, 128, 128, stream);
  run_layer(128, Ws,             bs,       Wd,             bd,       attn,       h, 128);
  conv_row(h, Ab, N, 128, 128, stream);
  run_layer(128, Ws + 128 * 128, bs + 128, Wd + 128 * 128, bd + 128, attn + 128, h, 128);

  // z2 = relu(hg @ lin_w + lin_b)  (WMMA), then tiny classifier
  conv_row(hg, Ab, B, 128, 128, stream);
  conv_tr(lin_w, Wt1, 128, 64, 128, stream);
  gemm(Ab, Wt1, lin_b, z2, B, 128, 64, 1, stream);
  hipLaunchKernelGGL(k_cls, dim3((B * 2 + TB - 1) / TB), dim3(TB), 0, stream,
                     z2, cls_w, cls_b, (float*)d_out, B);
}
